// IGAP_16879221473585
// MI455X (gfx1250) — matrix operand order repaired and checked by reading, not compiled, not run on hardware
//
#include <hip/hip_runtime.h>
#include <hip/hip_bf16.h>

// ---------------------------------------------------------------------------
// GraphSAGE (2x SAGEConv mean-agg) + MLP decoder + softmax, for gfx1250.
// GEMMs run on v_wmma_f32_16x16x32_bf16 (bf16 operands, f32 accumulate).
// ---------------------------------------------------------------------------

typedef __attribute__((ext_vector_type(8)))  float   v8f;
typedef __attribute__((ext_vector_type(8)))  __bf16  v8bf;
typedef __attribute__((ext_vector_type(16))) __bf16  v16bf;

#define NN 50000
#define EE 800000
#define DD 256

// ---------------------------------------------------------------- utilities
__global__ void IGAP_zero_f32(float* __restrict__ p, long long n) {
    long long i = (long long)blockIdx.x * blockDim.x + threadIdx.x;
    long long s = (long long)gridDim.x * blockDim.x;
    for (; i < n; i += s) p[i] = 0.0f;
}

__global__ void IGAP_cast_bf16(const float* __restrict__ src,
                               __bf16* __restrict__ dst, int n) {
    int i = blockIdx.x * blockDim.x + threadIdx.x;
    int s = gridDim.x * blockDim.x;
    for (; i < n; i += s) dst[i] = (__bf16)src[i];
}

// in-degree count (same graph for both layers -> computed once)
__global__ void IGAP_degree(const long long* __restrict__ ei,
                            float* __restrict__ cnt, int E) {
    int e = blockIdx.x * blockDim.x + threadIdx.x;
    int s = gridDim.x * blockDim.x;
    for (; e < E; e += s) {
        long long d = ei[(long long)E + e];
        atomicAdd(&cnt[d], 1.0f);
    }
}

__global__ void IGAP_invdeg(const float* __restrict__ cnt,
                            float* __restrict__ inv, int M) {
    int i = blockIdx.x * blockDim.x + threadIdx.x;
    if (i < M) inv[i] = 1.0f / fmaxf(cnt[i], 1.0f);
}

// one wave per edge: gather x[src] (2x float4 / lane) and scatter-add to agg[dst]
__global__ __launch_bounds__(256)
void IGAP_scatter_add(const float* __restrict__ x,
                      const long long* __restrict__ ei,
                      float* __restrict__ agg, int E) {
    int e = blockIdx.x * 8 + (threadIdx.x >> 5);
    if (e >= E) return;
    int lane = threadIdx.x & 31;
    long long s = ei[e];
    long long d = ei[(long long)E + e];
    const float* xs = x + s * DD + lane * 8;
    float4 a = *(const float4*)(xs);
    float4 b = *(const float4*)(xs + 4);
    float* ag = agg + d * DD + lane * 8;
    atomicAdd(ag + 0, a.x); atomicAdd(ag + 1, a.y);
    atomicAdd(ag + 2, a.z); atomicAdd(ag + 3, a.w);
    atomicAdd(ag + 4, b.x); atomicAdd(ag + 5, b.y);
    atomicAdd(ag + 6, b.z); atomicAdd(ag + 7, b.w);
}

// ------------------------------------------------------------ WMMA dual GEMM
// out[m,n] = relu( rs[m]*sum_k A1[m,k]*W1[n,k] + sum_k A2[m,k]*W2[n,k] + bias[n] )
// A* are f32 [M,256] row-major, W* are bf16 [256,256] row-major ([n][k]).
// Block: 256 threads = 8 waves (2 M x 4 N), block tile 64x128, wave tile 32x32.
__global__ __launch_bounds__(256, 2)
void IGAP_sage_gemm(const float* __restrict__ A1,
                    const float* __restrict__ rowscale,   // null -> 1.0
                    const __bf16* __restrict__ W1,
                    const float* __restrict__ A2,         // null -> single GEMM
                    const __bf16* __restrict__ W2,
                    const float* __restrict__ bias,
                    float* __restrict__ out, int M) {
    constexpr int K   = 256;
    constexpr int N   = 256;
    constexpr int BM  = 64;
    constexpr int LDA = 40;   // padded bf16 row stride (banks)

    __shared__ __attribute__((aligned(16))) __bf16 As[BM * LDA];

    const int tid   = threadIdx.x;
    const int lane  = tid & 31;
    const int wid   = tid >> 5;
    const int waveM = wid & 1;      // 0..1
    const int waveN = wid >> 1;     // 0..3

    const int m_blk = blockIdx.x * BM;
    const int n0    = blockIdx.y * 128 + waveN * 32;

    v8f acc[2][2];
#pragma unroll
    for (int i = 0; i < 2; ++i)
#pragma unroll
        for (int j = 0; j < 2; ++j)
#pragma unroll
            for (int v = 0; v < 8; ++v) acc[i][j][v] = 0.0f;

    const int lr = lane & 15;            // row-in-tile for A, n-in-tile for B
    const int kh = (lane >> 4) << 3;     // A frag k-half select {0,8}
    const int kb = (lane >> 4) << 4;     // unused; B frags select k half with kh

    // A staging coords: 64 rows x 32 cols, 8 elems/thread
    const int sr  = tid >> 2;            // 0..63
    const int scb = (tid & 3) << 3;      // 0,8,16,24
    int grow = m_blk + sr;
    if (grow >= M) grow = M - 1;         // clamp (stores are guarded later)

#pragma unroll 1
    for (int p = 0; p < 2; ++p) {
        const float*  A = (p == 0) ? A1 : A2;
        if (!A) break;
        const __bf16* W  = (p == 0) ? W1 : W2;
        const float*  sc = (p == 0) ? rowscale : nullptr;
        const float* aptr = A + (long long)grow * K + scb;
        const float  smul = sc ? sc[grow] : 1.0f;

        // software pipeline: f32 slab for step kk held in registers across the
        // WMMA phase of step kk-32, so only the ds_store sits between barriers.
        float4 f0 = *(const float4*)(aptr);
        float4 f1 = *(const float4*)(aptr + 4);

#pragma unroll 1
        for (int kk = 0; kk < K; kk += 32) {
            __syncthreads();   // previous iteration's ds_loads are done
            v8bf bv;
            bv[0] = (__bf16)(f0.x * smul); bv[1] = (__bf16)(f0.y * smul);
            bv[2] = (__bf16)(f0.z * smul); bv[3] = (__bf16)(f0.w * smul);
            bv[4] = (__bf16)(f1.x * smul); bv[5] = (__bf16)(f1.y * smul);
            bv[6] = (__bf16)(f1.z * smul); bv[7] = (__bf16)(f1.w * smul);
            *(v8bf*)(&As[sr * LDA + scb]) = bv;

            // issue next slab's global loads early (clamped on last step)
            int knext = (kk + 32 < K) ? (kk + 32) : kk;
            f0 = *(const float4*)(aptr + knext);
            f1 = *(const float4*)(aptr + knext + 4);
            if (kk + 32 < K) __builtin_prefetch(aptr + knext, 0, 1);

            __syncthreads();

            // A fragments (16x32 bf16): lanes 0-15 K=0..7 & 16..23; lanes 16-31 K=8..15 & 24..31
            v16bf af[2];
#pragma unroll
            for (int mt = 0; mt < 2; ++mt) {
                const __bf16* ab = &As[(waveM * 32 + mt * 16 + lr) * LDA + kh];
                v8bf lo = *(const v8bf*)(ab);
                v8bf hi = *(const v8bf*)(ab + 16);
#pragma unroll
                for (int q = 0; q < 8; ++q) { af[mt][q] = lo[q]; af[mt][8 + q] = hi[q]; }
            }

            // B fragments (32x16 bf16): lane n=lr, lanes 0-15 K=0..7 & 16..23; lanes 16-31 K=8..15 & 24..31
#pragma unroll
            for (int nt = 0; nt < 2; ++nt) {
                const __bf16* wb = W + (long long)(n0 + nt * 16 + lr) * K + kk + kh;
                v8bf lo = *(const v8bf*)(wb);
                v8bf hi = *(const v8bf*)(wb + 16);
                v16bf bf;
#pragma unroll
                for (int q = 0; q < 8; ++q) { bf[q] = lo[q]; bf[8 + q] = hi[q]; }
#pragma unroll
                for (int mt = 0; mt < 2; ++mt)
                    acc[mt][nt] = __builtin_amdgcn_wmma_f32_16x16x32_bf16(
                        false, af[mt], false, bf, (short)0, acc[mt][nt],
                        false, false);
            }
        }
    }

    // epilogue: C layout -> lane L holds N = L%16, rows M = v + 8*(L/16)
    const int mrow0 = m_blk + waveM * 32 + ((lane >> 4) << 3);
    const bool full = (m_blk + BM) <= M;   // 781/782 blocks take the fast path
#pragma unroll
    for (int mt = 0; mt < 2; ++mt)
#pragma unroll
        for (int nt = 0; nt < 2; ++nt) {
            int n = n0 + nt * 16 + lr;
            float bvv = bias[n];
            float* op = out + (long long)(mrow0 + mt * 16) * N + n;
            if (full) {
#pragma unroll
                for (int v = 0; v < 8; ++v)
                    op[(long long)v * N] = fmaxf(acc[mt][nt][v] + bvv, 0.0f);
            } else {
#pragma unroll
                for (int v = 0; v < 8; ++v) {
                    int m = mrow0 + mt * 16 + v;
                    if (m < M)
                        op[(long long)v * N] = fmaxf(acc[mt][nt][v] + bvv, 0.0f);
                }
            }
        }
}

// ---------------------------------------------- decoder tail: 256->8 + softmax
__global__ __launch_bounds__(256)
void IGAP_mlp2_softmax(const float* __restrict__ h,
                       const float* __restrict__ W,   // [8][256] f32
                       const float* __restrict__ b,   // [8]
                       float* __restrict__ out, int M) {
    int m = blockIdx.x * 8 + (threadIdx.x >> 5);
    if (m >= M) return;
    int lane = threadIdx.x & 31;
    const float* hr = h + (long long)m * DD;
    float accv[8];
#pragma unroll
    for (int o = 0; o < 8; ++o) accv[o] = 0.0f;
    for (int k = lane; k < DD; k += 32) {
        float hv = hr[k];
#pragma unroll
        for (int o = 0; o < 8; ++o) accv[o] = fmaf(hv, W[o * DD + k], accv[o]);
    }
#pragma unroll
    for (int o = 0; o < 8; ++o)
        for (int off = 16; off > 0; off >>= 1)
            accv[o] += __shfl_xor(accv[o], off);
    float logits[8], mx = -3.4e38f;
#pragma unroll
    for (int o = 0; o < 8; ++o) { logits[o] = accv[o] + b[o]; mx = fmaxf(mx, logits[o]); }
    float sum = 0.0f;
#pragma unroll
    for (int o = 0; o < 8; ++o) { logits[o] = __expf(logits[o] - mx); sum += logits[o]; }
    float inv = 1.0f / sum;
    if (lane < 8) out[(long long)m * 8 + lane] = logits[lane] * inv;
}

// ---------------------------------------------------------------------------
extern "C" void kernel_launch(void* const* d_in, const int* in_sizes, int n_in,
                              void* d_out, int out_size, void* d_ws, size_t ws_size,
                              hipStream_t stream) {
    (void)in_sizes; (void)n_in; (void)out_size; (void)ws_size;
    const int M = NN, E = EE;

    const float*     x    = (const float*)d_in[0];
    const long long* ei   = (const long long*)d_in[1];
    const float*     W1lf = (const float*)d_in[2];
    const float*     b1   = (const float*)d_in[3];
    const float*     W1rf = (const float*)d_in[4];
    const float*     W2lf = (const float*)d_in[5];
    const float*     b2   = (const float*)d_in[6];
    const float*     W2rf = (const float*)d_in[7];
    const float*     Wm1f = (const float*)d_in[8];
    const float*     bm1  = (const float*)d_in[9];
    const float*     Wm2  = (const float*)d_in[10];
    const float*     bm2  = (const float*)d_in[11];
    float* outp = (float*)d_out;

    // workspace carve-up
    size_t off = 0;
    char* base = (char*)d_ws;
    auto carve = [&](size_t bytes) {
        void* p = base + off;
        off = (off + bytes + 255) & ~(size_t)255;
        return p;
    };
    float*  agg  = (float*)carve((size_t)M * DD * 4);
    float*  bufA = (float*)carve((size_t)M * DD * 4);
    float*  bufB = (float*)carve((size_t)M * DD * 4);
    float*  cnt  = (float*)carve((size_t)M * 4);
    float*  invc = (float*)carve((size_t)M * 4);
    __bf16* w1l  = (__bf16*)carve((size_t)DD * DD * 2);
    __bf16* w1r  = (__bf16*)carve((size_t)DD * DD * 2);
    __bf16* w2l  = (__bf16*)carve((size_t)DD * DD * 2);
    __bf16* w2r  = (__bf16*)carve((size_t)DD * DD * 2);
    __bf16* wm1  = (__bf16*)carve((size_t)DD * DD * 2);

    dim3 blk(256);
    dim3 gemm_grid((M + 63) / 64, 2);

    // init + weight casts
    IGAP_zero_f32<<<2048, blk, 0, stream>>>(agg, (long long)M * DD);
    IGAP_zero_f32<<<64, blk, 0, stream>>>(cnt, M);
    IGAP_cast_bf16<<<128, blk, 0, stream>>>(W1lf, w1l, DD * DD);
    IGAP_cast_bf16<<<128, blk, 0, stream>>>(W1rf, w1r, DD * DD);
    IGAP_cast_bf16<<<128, blk, 0, stream>>>(W2lf, w2l, DD * DD);
    IGAP_cast_bf16<<<128, blk, 0, stream>>>(W2rf, w2r, DD * DD);
    IGAP_cast_bf16<<<128, blk, 0, stream>>>(Wm1f, wm1, DD * DD);

    // degrees (shared by both layers)
    IGAP_degree<<<1024, blk, 0, stream>>>(ei, cnt, E);
    IGAP_invdeg<<<(M + 255) / 256, blk, 0, stream>>>(cnt, invc, M);

    // layer 1: aggregate x, then h1 = relu(mean@W1l^T + x@W1r^T + b1)
    IGAP_scatter_add<<<(E + 7) / 8, blk, 0, stream>>>(x, ei, agg, E);
    IGAP_sage_gemm<<<gemm_grid, blk, 0, stream>>>(agg, invc, w1l, x, w1r, b1, bufA, M);

    // layer 2: aggregate h1, then h2 = relu(mean@W2l^T + h1@W2r^T + b2)
    IGAP_zero_f32<<<2048, blk, 0, stream>>>(agg, (long long)M * DD);
    IGAP_scatter_add<<<(E + 7) / 8, blk, 0, stream>>>(bufA, ei, agg, E);
    IGAP_sage_gemm<<<gemm_grid, blk, 0, stream>>>(agg, invc, w2l, bufA, w2r, b2, bufB, M);

    // MLP layer 1: h3 = relu(h2@Wm1^T + bm1)
    IGAP_sage_gemm<<<gemm_grid, blk, 0, stream>>>(bufB, nullptr, wm1, nullptr, nullptr,
                                                  bm1, bufA, M);

    // MLP layer 2 + softmax
    IGAP_mlp2_softmax<<<(M + 7) / 8, blk, 0, stream>>>(bufA, Wm2, bm2, outp, M);
}
